// SiameseKPConv_81518479278781
// MI455X (gfx1250) — matrix operand order repaired and checked by reading, not compiled, not run on hardware
//
#include <hip/hip_runtime.h>
#include <math.h>

// ---------------------------------------------------------------------------
// Siamese KPConv for MI455X (gfx1250, wave32).
//  * KNN: VALU register top-k (selection-bound, fp32-exact), batch mask
//    exploited structurally (two contiguous 4096-point halves).
//  * KPConv blocks: fused gather + aggregate + GEMM kernel using
//    v_wmma_f32_16x16x32_f16. Weights are pre-swizzled into WMMA fragment
//    order (K zero-padded to 32), streamed into LDS with double-buffered
//    GLOBAL_LOAD_ASYNC_TO_LDS_B128 copies pipelined against the VALU
//    aggregate stage and the WMMA issues; B fragments are single v16h reads.
//  * BN: deterministic fixed-order tree reductions (graph-replay safe).
// Workspace requirement: ~92 MB.
// ---------------------------------------------------------------------------

typedef __attribute__((ext_vector_type(16))) _Float16 v16h;
typedef __attribute__((ext_vector_type(4)))  _Float16 v4h;
typedef __attribute__((ext_vector_type(8)))  float    v8f;

#define NPTS   8192
#define HALF_N 4096
#define KNN    16
#define NKP    25
#define SIGMA_ 0.5f
#define BN_EPS_ 1e-5f

// ---------------------------------------------------------------------------
// KNN (k=16) within the same batch half; one thread per query, LDS-tiled
// support points, register-resident sorted insertion (ascending distances).
// ---------------------------------------------------------------------------
#define KTILE 256
__global__ __launch_bounds__(KTILE)
void knn16_kernel(const float* __restrict__ pos,
                  int* __restrict__ oidx, float* __restrict__ odist)
{
    __shared__ float sp[KTILE * 3];
    const int q  = blockIdx.x * KTILE + threadIdx.x;       // 8192 queries exactly
    const int s0 = (q < HALF_N) ? 0 : HALF_N;              // same for whole block
    const float qx = pos[q*3+0], qy = pos[q*3+1], qz = pos[q*3+2];

    float bd[KNN]; int bi[KNN];
#pragma unroll
    for (int i = 0; i < KNN; ++i) { bd[i] = 3.0e38f; bi[i] = 0; }

    for (int t = 0; t < HALF_N; t += KTILE) {
        __syncthreads();
        const int sI = s0 + t + threadIdx.x;
        sp[threadIdx.x*3+0] = pos[sI*3+0];
        sp[threadIdx.x*3+1] = pos[sI*3+1];
        sp[threadIdx.x*3+2] = pos[sI*3+2];
        __syncthreads();
        for (int j = 0; j < KTILE; ++j) {
            const float dx = qx - sp[j*3+0];
            const float dy = qy - sp[j*3+1];
            const float dz = qz - sp[j*3+2];
            float cd = dx*dx + dy*dy + dz*dz;
            if (cd < bd[KNN-1]) {                 // common-case skip
                int ci = s0 + t + j;
#pragma unroll
                for (int r = 0; r < KNN; ++r) {   // fully unrolled swap-chain
                    if (cd < bd[r]) {
                        float td = bd[r]; int ti = bi[r];
                        bd[r] = cd; bi[r] = ci;
                        cd = td; ci = ti;
                    }
                }
            }
        }
    }
#pragma unroll
    for (int i = 0; i < KNN; ++i) {
        oidx[q*KNN+i]  = bi[i];
        odist[q*KNN+i] = bd[i];                   // >= 0 by construction
    }
}

// 1-NN of pos2 queries against pos1 support (same half)
__global__ __launch_bounds__(KTILE)
void nn1_kernel(const float* __restrict__ qpos, const float* __restrict__ spos,
                int* __restrict__ onn)
{
    __shared__ float sp[KTILE * 3];
    const int q  = blockIdx.x * KTILE + threadIdx.x;
    const int s0 = (q < HALF_N) ? 0 : HALF_N;
    const float qx = qpos[q*3+0], qy = qpos[q*3+1], qz = qpos[q*3+2];
    float bestd = 3.0e38f; int besti = 0;

    for (int t = 0; t < HALF_N; t += KTILE) {
        __syncthreads();
        const int sI = s0 + t + threadIdx.x;
        sp[threadIdx.x*3+0] = spos[sI*3+0];
        sp[threadIdx.x*3+1] = spos[sI*3+1];
        sp[threadIdx.x*3+2] = spos[sI*3+2];
        __syncthreads();
        for (int j = 0; j < KTILE; ++j) {
            const float dx = qx - sp[j*3+0];
            const float dy = qy - sp[j*3+1];
            const float dz = qz - sp[j*3+2];
            const float d = dx*dx + dy*dy + dz*dz;
            if (d < bestd) { bestd = d; besti = s0 + t + j; }
        }
    }
    onn[q] = besti;
}

// ---------------------------------------------------------------------------
// Influence tensor (f16): infl[n][k][p] = relu(1 - |rel(n,k) - kp[p]| / sigma).
// ---------------------------------------------------------------------------
__global__ __launch_bounds__(256)
void influence_kernel(const float* __restrict__ pos, const int* __restrict__ idx,
                      const float* __restrict__ kp, _Float16* __restrict__ infl)
{
    __shared__ float skp[NKP * 3];
    if (threadIdx.x < NKP * 3) skp[threadIdx.x] = kp[threadIdx.x];
    __syncthreads();
    const int t = blockIdx.x * 256 + threadIdx.x;      // n*16 + k, 131072 total
    const int n = t >> 4;
    const int j = idx[t];
    const float rx = pos[j*3+0] - pos[n*3+0];
    const float ry = pos[j*3+1] - pos[n*3+1];
    const float rz = pos[j*3+2] - pos[n*3+2];
#pragma unroll
    for (int p = 0; p < NKP; ++p) {
        const float dx = rx - skp[p*3+0];
        const float dy = ry - skp[p*3+1];
        const float dz = rz - skp[p*3+2];
        const float s = sqrtf(dx*dx + dy*dy + dz*dz);
        infl[(size_t)t * NKP + p] = (_Float16)fmaxf(0.0f, 1.0f - s * (1.0f / SIGMA_));
    }
}

// ---------------------------------------------------------------------------
// Weight pre-swizzle: f32 [25][C][O] -> f16 fragment-ordered blocks of 512:
//   frag = (p*KPAD32 + cb32)*nColBlk + q ; element (lane, j):
//   K = cb32*32 + (lane<16 ? 0 : 16) + j (zero-padded past C), col = q*16+(lane&15)
// Makes each lane's WMMA B fragment one contiguous 32-byte LDS read, and the
// per-(p,cb) tile one contiguous 64*O-byte async copy. K-padding baked in.
// ---------------------------------------------------------------------------
__global__ __launch_bounds__(256)
void w_swizzle_kernel(const float* __restrict__ W, _Float16* __restrict__ out,
                      int C, int O, int KPAD, int total)
{
    const int i = blockIdx.x * 256 + threadIdx.x;
    if (i >= total) return;
    const int nColBlk = O >> 4;
    const int KPAD32  = KPAD >> 5;
    const int r    = i & 511;
    const int frag = i >> 9;
    const int lane = r >> 4;
    const int j    = r & 15;
    const int q    = frag % nColBlk;
    const int t2   = frag / nColBlk;
    const int cb32 = t2 % KPAD32;
    const int p    = t2 / KPAD32;
    const int K    = (cb32 << 5) + ((lane & 16) ? 16 : 0) + j;
    const int col  = (q << 4) + (lane & 15);
    out[i] = (K < C) ? (_Float16)W[((size_t)p * C + K) * O + col] : (_Float16)0.0f;
}

// ---------------------------------------------------------------------------
// Fused KPConv block: out[n,o] = sum_{p,c} (sum_k infl[n,k,p]*x[idx[n,k],c]) * W[p,c,o]
// One workgroup (4 waves / 128 thr) per 16-point row tile. Single path:
// fragment-ordered W tiles double-buffered via async global->LDS B128 copies,
// pipelined against the VALU aggregate stage and 4 unrolled WMMA issues.
// ---------------------------------------------------------------------------
__global__ __launch_bounds__(128)
void kpconv_wmma_kernel(const float* __restrict__ x, const int* __restrict__ idx,
                        const _Float16* __restrict__ infl, const _Float16* __restrict__ Whs,
                        float* __restrict__ out, int C, int O)
{
    __shared__ __align__(16) _Float16 xg[16*16*32];      // 16 KB  gathered feats (f16)
    __shared__ _Float16 fin[16*16*NKP];                  // 12.8 KB influence tile
    __shared__ __align__(32) _Float16 afrag[512];        // 1 KB  A fragment
    __shared__ __align__(32) _Float16 btile[2][32*256];  // 32 KB  W tiles (dbl-buf)

    const int tid  = threadIdx.x;
    const int lane = tid & 31;
    const int wave = tid >> 5;
    const int n0   = blockIdx.x * 16;
    const int nColBlk = O >> 4;
    const int KPAD   = (C + 31) & ~31;
    const int KPAD32 = KPAD >> 5;
    const int nChunk = (64 * O) >> 4;          // b128 transfers per 32xO f16 tile
    const int tileHalves = 32 * O;             // fragment-ordered tile size

    for (int i = tid; i < 16*16*NKP; i += 128)
        fin[i] = infl[(size_t)n0 * (16*NKP) + i];

    // A-fragment thread decomposition: m = tid>>3, 4 consecutive kk per thread.
    // The 4-kk group never straddles a fragment region -> constant lane map.
    const int am   = tid >> 3;                 // 0..15
    const int akk4 = (tid & 7) << 2;           // 0,4,...,28
    const int areg = akk4 >> 3;                // 0..3
    const int afl  = am + ((areg >= 2) ? 16 : 0);
    const int afj  = akk4 & 15;

    v8f acc[4];
#pragma unroll
    for (int b = 0; b < 4; ++b)
#pragma unroll
        for (int i = 0; i < 8; ++i) acc[b][i] = 0.0f;

    __syncthreads();

    for (int cb = 0; cb < KPAD; cb += 32) {
        const int cb32 = cb >> 5;
        // ---- prologue: async W tile for p=0 overlaps the feature gather
        {
            const char* ga = (const char*)(Whs + (size_t)cb32 * nColBlk * 512);
            char* lb = (char*)&btile[0][0];
            for (int e = tid; e < nChunk; e += 128) {
                const unsigned ldsa = (unsigned)(size_t)(lb + e * 16);
                asm volatile("global_load_async_to_lds_b128 %0, %1, off"
                             :: "v"(ldsa), "v"(ga + e * 16) : "memory");
            }
        }
        // ---- gather 16 points x 16 neighbors x 32 channels (f32 -> f16)
        for (int s = tid; s < 256; s += 128) {
            const int m = s >> 4, k = s & 15;
            const int src = idx[(n0 + m) * 16 + k];
            const float* xr = x + (size_t)src * C + cb;
            _Float16* dst = &xg[s * 32];
#pragma unroll 8
            for (int c = 0; c < 32; ++c)
                dst[c] = (_Float16)((cb + c < C) ? xr[c] : 0.0f);
        }
        __syncthreads();     // xg + fin visible (btile waited per-p below)

        for (int p = 0; p < NKP; ++p) {
            // ---- A fragment: agg[m][kk] = sum_k fin[m,k,p] * xg[m,k,kk]
            {
                float a0 = 0.0f, a1 = 0.0f, a2 = 0.0f, a3 = 0.0f;
#pragma unroll
                for (int k = 0; k < 16; ++k) {
                    const float w = (float)fin[(am*16 + k)*NKP + p];
                    const v4h xv = *(const v4h*)&xg[(am*16 + k)*32 + akk4];
                    a0 += w * (float)xv[0];
                    a1 += w * (float)xv[1];
                    a2 += w * (float)xv[2];
                    a3 += w * (float)xv[3];
                }
                v4h av;
                av[0] = (_Float16)a0; av[1] = (_Float16)a1;
                av[2] = (_Float16)a2; av[3] = (_Float16)a3;
                *(v4h*)&afrag[afl * 16 + afj] = av;
            }
            asm volatile("s_wait_asynccnt 0x0" ::: "memory");  // cur W tile done
            __syncthreads();

            const v16h aF = *(const v16h*)&afrag[lane * 16];
            const _Float16* bt = &btile[p & 1][0];
#pragma unroll
            for (int b = 0; b < 4; ++b) {
                const int q = wave + 4 * b;    // wave-uniform guard: EXEC stays full
                if (q < nColBlk) {
                    const v16h bF = *(const v16h*)&bt[q * 512 + lane * 16];
                    acc[b] = __builtin_amdgcn_wmma_f32_16x16x32_f16(
                                 false, aF, false, bF, (short)0, acc[b], false, false);
                }
            }
            // ---- issue async copy of next kernel point's W tile
            if (p + 1 < NKP) {
                const char* ga = (const char*)(Whs +
                    ((size_t)(p + 1) * KPAD32 + cb32) * nColBlk * 512);
                char* lb = (char*)&btile[(p + 1) & 1][0];
                for (int e = tid; e < nChunk; e += 128) {
                    const unsigned ldsa = (unsigned)(size_t)(lb + e * 16);
                    asm volatile("global_load_async_to_lds_b128 %0, %1, off"
                                 :: "v"(ldsa), "v"(ga + e * 16) : "memory");
                }
            }
            __syncthreads();
        }
    }

    // ---- write D tiles (16x16 f32 C/D layout, ISA 7.12.2)
    const int mb = (lane < 16) ? 0 : 8;
    const int cl = lane & 15;
#pragma unroll
    for (int b = 0; b < 4; ++b) {
        const int q = wave + 4 * b;
        if (q < nColBlk) {
#pragma unroll
            for (int r = 0; r < 8; ++r)
                out[(size_t)(n0 + mb + r) * O + q * 16 + cl] = acc[b][r];
        }
    }
}

// ---------------------------------------------------------------------------
// BatchNorm statistics: one block per channel, fixed-order tree reduction.
// ---------------------------------------------------------------------------
__global__ __launch_bounds__(256)
void bn_stats_kernel(const float* __restrict__ x, float* __restrict__ stats, int O)
{
    const int o = blockIdx.x;
    float s = 0.0f, s2 = 0.0f;
    for (int n = threadIdx.x; n < NPTS; n += 256) {
        const float v = x[(size_t)n * O + o];
        s += v; s2 += v * v;
    }
    __shared__ float sh[256], sh2[256];
    sh[threadIdx.x] = s; sh2[threadIdx.x] = s2;
    __syncthreads();
    for (int st = 128; st > 0; st >>= 1) {
        if (threadIdx.x < st) {
            sh[threadIdx.x]  += sh[threadIdx.x + st];
            sh2[threadIdx.x] += sh2[threadIdx.x + st];
        }
        __syncthreads();
    }
    if (threadIdx.x == 0) {
        const float m   = sh[0] * (1.0f / NPTS);
        const float var = sh2[0] * (1.0f / NPTS) - m * m;
        stats[o*2+0] = m;
        stats[o*2+1] = rsqrtf(var + BN_EPS_);
    }
}

__global__ __launch_bounds__(256)
void bn_act_kernel(float* __restrict__ x, const float* __restrict__ stats,
                   const float* __restrict__ g, const float* __restrict__ b,
                   float slope, int O, int total)
{
    const int i = blockIdx.x * 256 + threadIdx.x;
    if (i >= total) return;
    const int o = i % O;
    const float v = (x[i] - stats[o*2]) * stats[o*2+1] * g[o] + b[o];
    x[i] = (v >= 0.0f) ? v : slope * v;
}

// ---------------------------------------------------------------------------
// Small utility kernels
// ---------------------------------------------------------------------------
__global__ __launch_bounds__(256)
void concat_kernel(const float* __restrict__ a, int Ca,
                   const float* __restrict__ b, int Cb,
                   float* __restrict__ out, int total)
{
    const int i = blockIdx.x * 256 + threadIdx.x;
    if (i >= total) return;
    const int Co = Ca + Cb;
    const int n = i / Co, c = i - n * Co;
    out[i] = (c < Ca) ? a[(size_t)n * Ca + c] : b[(size_t)n * Cb + (c - Ca)];
}

__global__ __launch_bounds__(256)
void diff_kernel(const float* __restrict__ x2, const float* __restrict__ x1,
                 const int* __restrict__ nn1, float* __restrict__ out, int C, int total)
{
    const int i = blockIdx.x * 256 + threadIdx.x;
    if (i >= total) return;
    const int n = i / C, c = i - n * C;
    out[i] = x2[i] - x1[(size_t)nn1[n] * C + c];
}

// 3-NN inverse-squared-distance interpolation (uses first 3 of the sorted k=16)
__global__ __launch_bounds__(256)
void interp_kernel(const float* __restrict__ x, const int* __restrict__ idx16,
                   const float* __restrict__ d16, float* __restrict__ out, int C)
{
    const int n = blockIdx.x * 256 + threadIdx.x;
    const int i0 = idx16[n*16+0], i1 = idx16[n*16+1], i2 = idx16[n*16+2];
    float w0 = 1.0f / (d16[n*16+0] + 1e-8f);
    float w1 = 1.0f / (d16[n*16+1] + 1e-8f);
    float w2 = 1.0f / (d16[n*16+2] + 1e-8f);
    const float ws = 1.0f / (w0 + w1 + w2);
    w0 *= ws; w1 *= ws; w2 *= ws;
    for (int c = 0; c < C; ++c)
        out[(size_t)n*C + c] = w0 * x[(size_t)i0*C + c]
                             + w1 * x[(size_t)i1*C + c]
                             + w2 * x[(size_t)i2*C + c];
}

// plain GEMM for the tiny FC layers (16M / 1M MACs: negligible)
__global__ __launch_bounds__(256)
void fc_kernel(const float* __restrict__ x, const float* __restrict__ W,
               float* __restrict__ out, int C, int O)
{
    const int t = blockIdx.x * 256 + threadIdx.x;   // n*O + o
    const int n = t / O, o = t - n * O;
    float a = 0.0f;
    for (int c = 0; c < C; ++c) a += x[(size_t)n*C + c] * W[c*O + o];
    out[(size_t)n*O + o] = a;
}

// ---------------------------------------------------------------------------
// Orchestration
// ---------------------------------------------------------------------------
extern "C" void kernel_launch(void* const* d_in, const int* in_sizes, int n_in,
                              void* d_out, int out_size, void* d_ws, size_t ws_size,
                              hipStream_t stream)
{
    (void)in_sizes; (void)n_in; (void)out_size; (void)ws_size;

    const float* pos1  = (const float*)d_in[0];
    const float* feat1 = (const float*)d_in[1];
    const float* pos2  = (const float*)d_in[3];
    const float* feat2 = (const float*)d_in[4];
    const float* kp    = (const float*)d_in[7];
    const float* down_W[4] = { (const float*)d_in[8],  (const float*)d_in[9],
                               (const float*)d_in[10], (const float*)d_in[11] };
    const float* down_g[4] = { (const float*)d_in[12], (const float*)d_in[13],
                               (const float*)d_in[14], (const float*)d_in[15] };
    const float* down_b[4] = { (const float*)d_in[16], (const float*)d_in[17],
                               (const float*)d_in[18], (const float*)d_in[19] };
    const float* up_W[3]   = { (const float*)d_in[20], (const float*)d_in[21],
                               (const float*)d_in[22] };
    const float* up_g[3]   = { (const float*)d_in[23], (const float*)d_in[24],
                               (const float*)d_in[25] };
    const float* up_b[3]   = { (const float*)d_in[26], (const float*)d_in[27],
                               (const float*)d_in[28] };
    const float* fc1_W = (const float*)d_in[29];
    const float* fc_g  = (const float*)d_in[30];
    const float* fc_b  = (const float*)d_in[31];
    const float* fc2_W = (const float*)d_in[32];

    // ---- workspace carving (bump allocator, 256B aligned) ----
    char* ws = (char*)d_ws;
    size_t off = 0;
    auto carve = [&](size_t bytes) -> void* {
        void* p = ws + off;
        off += (bytes + 255) & ~(size_t)255;
        return p;
    };
    int*      idx16_1 = (int*)     carve((size_t)NPTS*16*4);
    float*    d16_1   = (float*)   carve((size_t)NPTS*16*4);
    int*      idx16_2 = (int*)     carve((size_t)NPTS*16*4);
    float*    d16_2   = (float*)   carve((size_t)NPTS*16*4);
    int*      nn1i    = (int*)     carve((size_t)NPTS*4);
    _Float16* infl1   = (_Float16*)carve((size_t)NPTS*16*NKP*2);  // 6.5 MB
    _Float16* infl2   = (_Float16*)carve((size_t)NPTS*16*NKP*2);
    float*    x1A     = (float*)   carve((size_t)NPTS*256*4);
    float*    x1B     = (float*)   carve((size_t)NPTS*256*4);
    float*    x2A     = (float*)   carve((size_t)NPTS*256*4);
    float*    x2B     = (float*)   carve((size_t)NPTS*256*4);
    float*    skip0   = (float*)   carve((size_t)NPTS*32*4);
    float*    skip1   = (float*)   carve((size_t)NPTS*64*4);
    float*    skip2   = (float*)   carve((size_t)NPTS*128*4);
    float*    xu0     = (float*)   carve((size_t)NPTS*256*4);
    float*    xu1     = (float*)   carve((size_t)NPTS*256*4);
    float*    xint    = (float*)   carve((size_t)NPTS*256*4);
    float*    xcat    = (float*)   carve((size_t)NPTS*384*4);
    float*    hbuf    = (float*)   carve((size_t)NPTS*64*4);
    _Float16* Wh      = (_Float16*)carve((size_t)NKP*384*128*2);  // 2.46 MB (max W)
    float*    stats   = (float*)   carve((size_t)256*2*4);

    const int EL = 256;   // elementwise block size
    auto grid1 = [](int total) { return (total + 255) / 256; };

    // ---- position-only precomputations (reused by every block) ----
    knn16_kernel<<<NPTS/KTILE, KTILE, 0, stream>>>(pos1, idx16_1, d16_1);
    knn16_kernel<<<NPTS/KTILE, KTILE, 0, stream>>>(pos2, idx16_2, d16_2);
    nn1_kernel  <<<NPTS/KTILE, KTILE, 0, stream>>>(pos2, pos1, nn1i);
    influence_kernel<<<NPTS*16/EL, EL, 0, stream>>>(pos1, idx16_1, kp, infl1);
    influence_kernel<<<NPTS*16/EL, EL, 0, stream>>>(pos2, idx16_2, kp, infl2);

    // ---- initial features: x = [pos, feat]  (C=6) ----
    concat_kernel<<<grid1(NPTS*6), EL, 0, stream>>>(pos1, 3, feat1, 3, x1A, NPTS*6);
    concat_kernel<<<grid1(NPTS*6), EL, 0, stream>>>(pos2, 3, feat2, 3, x2A, NPTS*6);

    // ---- down path (shared weights on both clouds) ----
    const int DCH[4] = {32, 64, 128, 256};
    float* x1cur = x1A; float* x1nxt = x1B;
    float* x2cur = x2A; float* x2nxt = x2B;
    float* skips[3] = {skip0, skip1, skip2};
    int Cin = 6;
    for (int i = 0; i < 4; ++i) {
        const int Cout = DCH[i];
        const int KPAD = (Cin + 31) & ~31;
        const int wlen = NKP * KPAD * Cout;
        w_swizzle_kernel<<<grid1(wlen), EL, 0, stream>>>(down_W[i], Wh, Cin, Cout, KPAD, wlen);

        kpconv_wmma_kernel<<<NPTS/16, 128, 0, stream>>>(x1cur, idx16_1, infl1, Wh, x1nxt, Cin, Cout);
        bn_stats_kernel<<<Cout, 256, 0, stream>>>(x1nxt, stats, Cout);
        bn_act_kernel<<<grid1(NPTS*Cout), EL, 0, stream>>>(x1nxt, stats, down_g[i], down_b[i], 0.1f, Cout, NPTS*Cout);

        kpconv_wmma_kernel<<<NPTS/16, 128, 0, stream>>>(x2cur, idx16_2, infl2, Wh, x2nxt, Cin, Cout);
        bn_stats_kernel<<<Cout, 256, 0, stream>>>(x2nxt, stats, Cout);
        bn_act_kernel<<<grid1(NPTS*Cout), EL, 0, stream>>>(x2nxt, stats, down_g[i], down_b[i], 0.1f, Cout, NPTS*Cout);

        if (i < 3)
            diff_kernel<<<grid1(NPTS*Cout), EL, 0, stream>>>(x2nxt, x1nxt, nn1i, skips[i], Cout, NPTS*Cout);

        float* t;
        t = x1cur; x1cur = x1nxt; x1nxt = t;
        t = x2cur; x2cur = x2nxt; x2nxt = t;
        Cin = Cout;
    }

    // ---- bottom: x = x2 - x1[nn1]   (C=256) ----
    diff_kernel<<<grid1(NPTS*256), EL, 0, stream>>>(x2cur, x1cur, nn1i, xu0, 256, NPTS*256);

    // ---- up path ----
    const int UCH_OUT[3] = {128, 64, 32};
    const int SKC[3]     = {128, 64, 32};    // skip channels used at up step i
    float* ucur = xu0; float* unxt = xu1;
    int Ccur = 256;
    for (int i = 0; i < 3; ++i) {
        const float* sk = (i == 0) ? skip2 : (i == 1) ? skip1 : skip0;
        const int Cinu = Ccur + SKC[i];
        const int Cout = UCH_OUT[i];

        interp_kernel<<<NPTS/EL, EL, 0, stream>>>(ucur, idx16_2, d16_2, xint, Ccur);
        concat_kernel<<<grid1(NPTS*Cinu), EL, 0, stream>>>(xint, Ccur, sk, SKC[i], xcat, NPTS*Cinu);

        const int KPAD = (Cinu + 31) & ~31;
        const int wlen = NKP * KPAD * Cout;
        w_swizzle_kernel<<<grid1(wlen), EL, 0, stream>>>(up_W[i], Wh, Cinu, Cout, KPAD, wlen);
        kpconv_wmma_kernel<<<NPTS/16, 128, 0, stream>>>(xcat, idx16_2, infl2, Wh, unxt, Cinu, Cout);
        bn_stats_kernel<<<Cout, 256, 0, stream>>>(unxt, stats, Cout);
        bn_act_kernel<<<grid1(NPTS*Cout), EL, 0, stream>>>(unxt, stats, up_g[i], up_b[i], 0.1f, Cout, NPTS*Cout);

        float* t = ucur; ucur = unxt; unxt = t;
        Ccur = Cout;
    }

    // ---- classifier head ----
    fc_kernel<<<grid1(NPTS*64), EL, 0, stream>>>(ucur, fc1_W, hbuf, 32, 64);
    bn_stats_kernel<<<64, 256, 0, stream>>>(hbuf, stats, 64);
    bn_act_kernel<<<grid1(NPTS*64), EL, 0, stream>>>(hbuf, stats, fc_g, fc_b, 0.2f, 64, NPTS*64);
    fc_kernel<<<grid1(NPTS*2), EL, 0, stream>>>(hbuf, fc2_W, (float*)d_out, 64, 2);
}
